// ShallowRGNNNet_71382356459697
// MI455X (gfx1250) — matrix operand order repaired and checked by reading, not compiled, not run on hardware
//
#include <hip/hip_runtime.h>
#include <cmath>

// ---------------- problem constants ----------------
#define B_    1024
#define N_    22
#define T_    1124
#define E_    200
#define KNL_  10
#define KS_   25
#define P_    20
#define TC_   1100     // T - KS + 1
#define TP_   55       // TC / P
#define HID_  20
#define OUT_  4
#define EPS_  1e-5f

#define ROWPAD_ 1136   // >= 68*16 + 15 + 16 + 15 + 1 = 1135, zero-padded past T_
#define NTILE_  69     // ceil(TC/16)
#define PW_     56     // pooled windows padded (55 real + 1 dump)

typedef __attribute__((ext_vector_type(16))) _Float16 v16h;
typedef __attribute__((ext_vector_type(8)))  float    v8f;
typedef __attribute__((ext_vector_type(2)))  float    v2f;

// ---------------- workspace layout (float offsets) ----------------
static constexpr size_t POOLED_OFF = 1024;
static constexpr size_t POOLED_SZ  = (size_t)B_ * KNL_ * N_ * TP_;      // 12,390,400
static constexpr size_t Z_OFF      = POOLED_OFF + POOLED_SZ;
static constexpr size_t ZH_SZ      = (size_t)B_ * KNL_ * N_ * HID_;     // 4,505,600
static constexpr size_t H_OFF      = Z_OFF + ZH_SZ;

// ---------------- helpers ----------------
// A-matrix (16x32 f16, MxK) lane layout per CDNA5 ISA §7.12.2:
//  lane L: M = L&15, half = L>>4; elem i: j=i>>1,p=i&1;
//  K = j<4 ? j*2+p+half*8 : 16+(j-4)*2+p+half*8
__device__ inline v16h load_conv_A(const float* __restrict__ conv_w, int lane) {
    int half = lane >> 4;
    int m    = lane & 15;
    v16h a;
    #pragma unroll
    for (int i = 0; i < 16; ++i) {
        int j = i >> 1, p = i & 1;
        int K = (j < 4) ? (j * 2 + p + half * 8) : (16 + (j - 4) * 2 + p + half * 8);
        float w = (m < KNL_ && K < KS_) ? conv_w[m * KS_ + K] : 0.0f;
        a[i] = (_Float16)w;
    }
    return a;
}

// B-matrix (32x16 f16, KxN) built from zero-padded LDS row:
//  lane L: col = L&15, K = i<8 ? i+(L>>4)*8 : 16+(i-8)+(L>>4)*8  ->  value = row[t0 + col + K]
__device__ inline v16h load_conv_B_lds(const float* __restrict__ row, int t0, int lane) {
    int tb = t0 + (lane & 15) + (lane >> 4) * 8;
    v16h b;
    #pragma unroll
    for (int i = 0; i < 16; ++i) b[i] = (_Float16)row[tb + i + ((i >= 8) ? 8 : 0)];
    return b;
}

// branchless ELU(alpha=1): max(v,0) + min(exp(v)-1, 0)
__device__ inline float elu1(float v) {
    return fmaxf(v, 0.0f) + fminf(__expf(v) - 1.0f, 0.0f);
}

// ---------------- k0: zero accumulators ----------------
__global__ void zero_kernel(float* __restrict__ ws) {
    for (int i = threadIdx.x; i < 548; i += blockDim.x) ws[i] = 0.0f;
}

// ---------------- k1: conv + ELU + BN statistics (WMMA f16) ----------------
// one block per (b,n) row; input row staged in zero-padded LDS
__global__ __launch_bounds__(256) void conv_stats_kernel(
    const float* __restrict__ in, const float* __restrict__ conv_w,
    const float* __restrict__ conv_b, float* __restrict__ stats) {
    __shared__ float row[ROWPAD_];
    __shared__ float ssum[16], ssq[16];
    int tid = threadIdx.x;
    if (tid < 16) { ssum[tid] = 0.0f; ssq[tid] = 0.0f; }

    int rowIdx = blockIdx.x;
    const float* src = in + (size_t)rowIdx * T_;
    for (int i = tid; i < ROWPAD_; i += 256) row[i] = (i < T_) ? src[i] : 0.0f;
    __syncthreads();

    int lane = tid & 31, wave = tid >> 5, half = lane >> 4, col = lane & 15;
    v16h a = load_conv_A(conv_w, lane);

    float kmask[8], bias[8], accS[8], accQ[8];
    #pragma unroll
    for (int r = 0; r < 8; ++r) {
        int k    = r + half * 8;
        kmask[r] = (k < KNL_) ? 1.0f : 0.0f;
        bias[r]  = conv_b[(k < KNL_) ? k : 0];
        accS[r]  = 0.0f;
        accQ[r]  = 0.0f;
    }

    for (int tile = wave; tile < NTILE_; tile += 8) {
        int t0 = tile * 16;
        v16h bm = load_conv_B_lds(row, t0, lane);
        v8f c = {};
        c = __builtin_amdgcn_wmma_f32_16x16x32_f16(false, a, false, bm,
                                                   (short)0, c, false, false);
        float tm = (t0 + col < TC_) ? 1.0f : 0.0f;
        #pragma unroll
        for (int r = 0; r < 8; ++r) {
            float v = elu1(c[r] + bias[r]);
            float m = kmask[r] * tm;
            accS[r] += m * v;
            accQ[r] += m * v * v;
        }
    }
    #pragma unroll
    for (int r = 0; r < 8; ++r) {
        int k = r + half * 8;
        atomicAdd(&ssum[k], accS[r]);
        atomicAdd(&ssq[k],  accQ[r]);
    }
    __syncthreads();
    if (tid < 16) {
        atomicAdd(&stats[tid],      ssum[tid]);
        atomicAdd(&stats[16 + tid], ssq[tid]);
    }
}

// ---------------- k2: BN finalize ----------------
__global__ void bn_finalize_kernel(const float* __restrict__ stats,
                                   const float* __restrict__ gamma,
                                   const float* __restrict__ beta,
                                   float* __restrict__ bn) {
    int k = threadIdx.x;
    if (k < 16) {
        float cnt  = (float)B_ * N_ * TC_;
        float mean = stats[k] / cnt;
        float var  = stats[16 + k] / cnt - mean * mean;
        float g    = (k < KNL_) ? gamma[k] : 0.0f;
        float be   = (k < KNL_) ? beta[k]  : 0.0f;
        float sc   = g * rsqrtf(var + EPS_);
        bn[k]      = sc;
        bn[16 + k] = be - mean * sc;
    }
}

// ---------------- k3: conv + ELU + BN + avgpool (WMMA f16) ----------------
// one block per (b,n) row; pool LDS padded 16x56 so all lanes add
// unconditionally (k>=10 rows and t>=TC column are dump slots with value 0,
// because bn scale/shift are 0 for k>=10 and are never read back).
__global__ __launch_bounds__(256) void conv_bn_pool_kernel(
    const float* __restrict__ in, const float* __restrict__ conv_w,
    const float* __restrict__ conv_b, const float* __restrict__ bn,
    float* __restrict__ pooled) {
    __shared__ float row[ROWPAD_];
    __shared__ float pool[16 * PW_];
    int tid = threadIdx.x;
    for (int i = tid; i < 16 * PW_; i += 256) pool[i] = 0.0f;

    int rowIdx = blockIdx.x;
    int b = rowIdx / N_, n = rowIdx % N_;
    const float* src = in + (size_t)rowIdx * T_;
    for (int i = tid; i < ROWPAD_; i += 256) row[i] = (i < T_) ? src[i] : 0.0f;
    __syncthreads();

    int lane = tid & 31, wave = tid >> 5, half = lane >> 4, col = lane & 15;
    v16h a = load_conv_A(conv_w, lane);

    float bias[8], scale[8], shift[8];
    #pragma unroll
    for (int r = 0; r < 8; ++r) {
        int k    = r + half * 8;
        bias[r]  = conv_b[(k < KNL_) ? k : 0];
        scale[r] = bn[k];           // 0 for k >= KNL_
        shift[r] = bn[16 + k];      // 0 for k >= KNL_
    }

    for (int tile = wave; tile < NTILE_; tile += 8) {
        int t0 = tile * 16;
        v16h bm = load_conv_B_lds(row, t0, lane);
        v8f c = {};
        c = __builtin_amdgcn_wmma_f32_16x16x32_f16(false, a, false, bm,
                                                   (short)0, c, false, false);
        int w = (t0 + col) / P_;    // 0..55 (55 = dump column)
        #pragma unroll
        for (int r = 0; r < 8; ++r) {
            int k   = r + half * 8;
            float v = elu1(c[r] + bias[r]) * scale[r] + shift[r];
            atomicAdd(&pool[k * PW_ + w], v);
        }
    }
    __syncthreads();
    const float inv = 1.0f / (float)P_;
    for (int i = tid; i < KNL_ * TP_; i += 256) {
        int k = i / TP_, w = i % TP_;
        pooled[(((size_t)b * KNL_ + k) * N_ + n) * TP_ + w] = pool[k * PW_ + w] * inv;
    }
}

// ---------------- k4: dense normalized adjacency (gcn_norm + self loops) ----
__global__ void ahat_kernel(const int* __restrict__ ei,
                            const float* __restrict__ ew,
                            float* __restrict__ Ahat) {
    __shared__ float deg[N_], dinv[N_];
    int tid = threadIdx.x;
    if (tid < N_) deg[tid] = 1.0f;               // self-loop weight 1
    __syncthreads();
    for (int e = tid; e < E_; e += blockDim.x)
        atomicAdd(&deg[ei[E_ + e]], ew[e]);      // col = dst
    __syncthreads();
    if (tid < N_) {
        float d = deg[tid];
        dinv[tid] = (d > 0.0f) ? rsqrtf(d) : 0.0f;
    }
    __syncthreads();
    for (int e = tid; e < E_; e += blockDim.x) {
        int r = ei[e], c = ei[E_ + e];
        atomicAdd(&Ahat[c * N_ + r], dinv[r] * ew[e] * dinv[c]);
    }
    if (tid < N_) atomicAdd(&Ahat[tid * N_ + tid], dinv[tid] * dinv[tid]);
}

// ---------------- k5: Z = X @ lin_w^T  (WMMA f32 16x16x4) ----------------
// block = 8 waves, handles 2 (b,k) pairs; zero-padded X (2 x 32x56) and
// lin_w (32x56) staged in LDS -> all K-step loads unconditional.
// A 16x4 f32: a[j] -> K = j + 2*(lane>>4), M = lane&15.  B mirrored.
__global__ __launch_bounds__(256) void lin_kernel(
    const float* __restrict__ pooled, const float* __restrict__ lin_w,
    float* __restrict__ Z) {
    __shared__ float Wp[32 * PW_];
    __shared__ float Xp[2][32 * PW_];
    int tid = threadIdx.x;
    size_t bk0 = (size_t)blockIdx.x * 2;
    for (int idx = tid; idx < 32 * PW_; idx += 256) {
        int r = idx / PW_, f = idx % PW_;
        bool inW = (r < HID_) && (f < TP_);
        bool inX = (r < N_)   && (f < TP_);
        Wp[idx]    = inW ? lin_w[r * TP_ + f] : 0.0f;
        Xp[0][idx] = inX ? pooled[(bk0 + 0) * N_ * TP_ + r * TP_ + f] : 0.0f;
        Xp[1][idx] = inX ? pooled[(bk0 + 1) * N_ * TP_ + r * TP_ + f] : 0.0f;
    }
    __syncthreads();

    int lane = tid & 31, wave = tid >> 5, half = lane >> 4;
    int p = wave >> 2, mt = (wave >> 1) & 1, nt = wave & 1;
    int node = mt * 16 + (lane & 15);
    int o    = nt * 16 + (lane & 15);
    const float* X = Xp[p];

    v8f c = {};
    #pragma unroll
    for (int ks = 0; ks < 14; ++ks) {
        int f0 = ks * 4 + 2 * half;       // padded col 55 is zero
        v2f a, bv;
        a[0]  = X[node * PW_ + f0];
        a[1]  = X[node * PW_ + f0 + 1];
        bv[0] = Wp[o * PW_ + f0];
        bv[1] = Wp[o * PW_ + f0 + 1];
        c = __builtin_amdgcn_wmma_f32_16x16x4_f32(false, a, false, bv,
                                                  (short)0, c, false, false);
    }
    size_t bk = bk0 + p;
    #pragma unroll
    for (int r = 0; r < 8; ++r) {
        int nn = mt * 16 + r + half * 8;
        if (nn < N_ && o < HID_)
            Z[(bk * N_ + nn) * HID_ + o] = c[r];
    }
}

// ---------------- k6: H = Ahat @ Z + lin_b (node mixing) ----------------
__global__ void agg_kernel(const float* __restrict__ Ahat,
                           const float* __restrict__ Z,
                           const float* __restrict__ lin_b,
                           float* __restrict__ H) {
    size_t idx = (size_t)blockIdx.x * blockDim.x + threadIdx.x;
    if (idx >= ZH_SZ) return;
    int    o  = (int)(idx % HID_);
    int    n  = (int)((idx / HID_) % N_);
    size_t bk = idx / ((size_t)HID_ * N_);
    const float* zb = Z + bk * N_ * HID_;
    float s = lin_b[o];
    #pragma unroll
    for (int m = 0; m < N_; ++m) s += Ahat[n * N_ + m] * zb[m * HID_ + o];
    H[idx] = s;   // same flat layout as reference h.reshape(B,-1)
}

// ---------------- k7: final FC [B,4400] -> [B,4] ----------------
__global__ void fc_kernel(const float* __restrict__ H,
                          const float* __restrict__ fc_w,
                          const float* __restrict__ fc_b,
                          float* __restrict__ out) {
    int idx = blockIdx.x * blockDim.x + threadIdx.x;
    if (idx >= B_ * OUT_) return;
    int b = idx >> 2, o = idx & 3;
    const int KF4 = (KNL_ * N_ * HID_) / 4;      // 1100
    const float4* h4 = (const float4*)(H + (size_t)b * KNL_ * N_ * HID_);
    const float4* w4 = (const float4*)(fc_w + (size_t)o * KNL_ * N_ * HID_);
    float s = fc_b[o];
    for (int i = 0; i < KF4; ++i) {
        float4 hv = h4[i], wv = w4[i];
        s += hv.x * wv.x + hv.y * wv.y + hv.z * wv.z + hv.w * wv.w;
    }
    out[idx] = s;
}

// ---------------- launcher ----------------
extern "C" void kernel_launch(void* const* d_in, const int* in_sizes, int n_in,
                              void* d_out, int out_size, void* d_ws, size_t ws_size,
                              hipStream_t stream) {
    (void)in_sizes; (void)n_in; (void)out_size; (void)ws_size;
    const float* in     = (const float*)d_in[0];
    const int*   ei     = (const int*)  d_in[1];
    const float* conv_w = (const float*)d_in[2];
    const float* conv_b = (const float*)d_in[3];
    const float* gamma  = (const float*)d_in[4];
    const float* beta   = (const float*)d_in[5];
    const float* ew     = (const float*)d_in[6];
    const float* lin_w  = (const float*)d_in[7];
    const float* lin_b  = (const float*)d_in[8];
    const float* fc_w   = (const float*)d_in[9];
    const float* fc_b   = (const float*)d_in[10];
    float* ws     = (float*)d_ws;
    float* stats  = ws;
    float* bn     = ws + 32;
    float* Ahat   = ws + 64;
    float* pooled = ws + POOLED_OFF;
    float* Z      = ws + Z_OFF;
    float* H      = ws + H_OFF;
    float* out    = (float*)d_out;

    zero_kernel<<<1, 256, 0, stream>>>(ws);
    conv_stats_kernel<<<B_ * N_, 256, 0, stream>>>(in, conv_w, conv_b, stats);
    bn_finalize_kernel<<<1, 16, 0, stream>>>(stats, gamma, beta, bn);
    conv_bn_pool_kernel<<<B_ * N_, 256, 0, stream>>>(in, conv_w, conv_b, bn, pooled);
    ahat_kernel<<<1, 256, 0, stream>>>(ei, ew, Ahat);
    lin_kernel<<<(B_ * KNL_) / 2, 256, 0, stream>>>(pooled, lin_w, Z);
    agg_kernel<<<(unsigned)((ZH_SZ + 255) / 256), 256, 0, stream>>>(Ahat, Z, lin_b, H);
    fc_kernel<<<(B_ * OUT_ + 255) / 256, 256, 0, stream>>>(H, fc_w, fc_b, out);
}
